// refinerfea_70540542869681
// MI455X (gfx1250) — matrix operand order repaired and checked by reading, not compiled, not run on hardware
//
#include <hip/hip_runtime.h>
#include <hip/hip_bf16.h>

#define BATCH 8
#define CH    64
#define NPT   2048

typedef __attribute__((ext_vector_type(16))) _Float16 v16h;
typedef __attribute__((ext_vector_type(8)))  _Float16 v8h;
typedef __attribute__((ext_vector_type(2)))  _Float16 v2h;
typedef __attribute__((ext_vector_type(8)))  float    v8f;
typedef __attribute__((ext_vector_type(4)))  float    v4f;
typedef __attribute__((ext_vector_type(2)))  float    v2f;

union V16 { v16h v; v8h h[2]; };

// ---------------------------------------------------------------------------
// Kernel 0: transpose feature [b][c][n] -> featT [b][n][c] so the KNN gather
// can use contiguous b128 row loads instead of 1024 strided b32 gathers.
// ---------------------------------------------------------------------------
__global__ __launch_bounds__(256) void k0_tr(const float* __restrict__ feature,
                                             float* __restrict__ featT) {
    __shared__ float t[64][65];                   // +1 pad: bank-conflict free
    const int b  = blockIdx.x >> 5;
    const int n0 = (blockIdx.x & 31) * 64;
    const int tx = threadIdx.x & 63;
    const int ty = threadIdx.x >> 6;              // 0..3

    const float* fb = feature + (size_t)b * CH * NPT;
#pragma unroll
    for (int c = ty; c < CH; c += 4) t[c][tx] = fb[(size_t)c * NPT + n0 + tx];
    __syncthreads();

    float* ob = featT + (size_t)b * NPT * CH + (size_t)n0 * CH;
#pragma unroll
    for (int n = ty; n < 64; n += 4) ob[(size_t)n * CH + tx] = t[tx][n];
}

// ---------------------------------------------------------------------------
// Kernel 1: KNN (top-17 smallest d2), inverse-distance weights, gather-max.
// One thread per query point; xyz staged in LDS; d2 scan vectorized 4-wide;
// neighbor rows gathered as contiguous v4f from featT.
// ---------------------------------------------------------------------------
__global__ __launch_bounds__(256) void k1_knn(const float* __restrict__ xyz,
                                              const float* __restrict__ featT,
                                              float* __restrict__ fea2) {
    __shared__ float sx[3 * NPT];                 // 24 KB
    const int b   = blockIdx.x >> 3;
    const int n0  = (blockIdx.x & 7) * 256;
    const int tid = threadIdx.x;

    const float* xb = xyz + (size_t)b * 3 * NPT;
    for (int i = tid; i < 3 * NPT; i += 256) sx[i] = xb[i];
    __syncthreads();

    const int n = n0 + tid;
    const float px = sx[n], py = sx[NPT + n], pz = sx[2 * NPT + n];

    float bd[17]; int bi[17];
#pragma unroll
    for (int j = 0; j < 17; ++j) { bd[j] = 3.4e38f; bi[j] = 0; }

    for (int m = 0; m < NPT; m += 4) {
        v4f mxv = *(const v4f*)(sx + m);
        v4f myv = *(const v4f*)(sx + NPT + m);
        v4f mzv = *(const v4f*)(sx + 2 * NPT + m);
#pragma unroll
        for (int e = 0; e < 4; ++e) {
            float dx = px - mxv[e];
            float dy = py - myv[e];
            float dz = pz - mzv[e];
            float d2 = dx * dx + dy * dy + dz * dz;
            if (d2 < bd[16]) {                    // register-resident sorted insert
                float v = d2; int vi = m + e;
#pragma unroll
                for (int j = 0; j < 17; ++j) {
                    bool sm = v < bd[j];
                    float tv = bd[j]; int ti = bi[j];
                    bd[j] = sm ? v  : tv;  bi[j] = sm ? vi : ti;
                    v     = sm ? tv : v;   vi    = sm ? ti : vi;
                }
            }
        }
    }

    // slots 1..16: nearest neighbors excluding self
    float w[16]; int idx[16]; float wsum = 0.f;
#pragma unroll
    for (int k = 0; k < 16; ++k) {
        float dis = sqrtf(fmaxf(bd[k + 1], 1e-12f));
        w[k] = 1.0f / dis;
        wsum += w[k];
        idx[k] = bi[k + 1];
    }
    const float inv = 1.0f / (wsum + 1e-7f);
#pragma unroll
    for (int k = 0; k < 16; ++k) w[k] *= inv;

    // gather-max over 16 neighbor rows (contiguous v4f loads)
    v4f mx4[16];
#pragma unroll
    for (int cc = 0; cc < 16; ++cc)
#pragma unroll
        for (int e = 0; e < 4; ++e) mx4[cc][e] = -3.4e38f;

    const float* ftb = featT + (size_t)b * NPT * CH;
#pragma unroll 1
    for (int k = 0; k < 16; ++k) {
        const float* fr = ftb + (size_t)idx[k] * CH;
        const float wk = w[k];
#pragma unroll
        for (int cc = 0; cc < 16; ++cc) {
            v4f fv = *(const v4f*)(fr + cc * 4);
#pragma unroll
            for (int e = 0; e < 4; ++e) mx4[cc][e] = fmaxf(mx4[cc][e], wk * fv[e]);
        }
    }

    float* ob = fea2 + (size_t)b * CH * NPT;
#pragma unroll
    for (int cc = 0; cc < 16; ++cc)
#pragma unroll
        for (int e = 0; e < 4; ++e) ob[(size_t)(cc * 4 + e) * NPT + n] = mx4[cc][e];
}

// ---------------------------------------------------------------------------
// Kernel 2: f,g (16-ch) and h (64-ch) projections -> f16 in WMMA-friendly
// layouts. gT/fT: [b][n][32] halves (c 0..15 real, 16..31 zero pad).
// h16: [b][c][2048] but with keys INTERLEAVED inside each 32-block
// (position 2j <-> key j, 2j+1 <-> key 16+j) so the attention kernel can
// store packed P pairs while keeping the PV B-operand loads contiguous.
// ---------------------------------------------------------------------------
__global__ __launch_bounds__(256) void k2_fgh(const float* __restrict__ fea2,
                                              const float* __restrict__ Wf, const float* __restrict__ bf,
                                              const float* __restrict__ Wg, const float* __restrict__ bg,
                                              const float* __restrict__ Wh, const float* __restrict__ bh,
                                              _Float16* __restrict__ gT,
                                              _Float16* __restrict__ fT,
                                              _Float16* __restrict__ h16) {
    __shared__ float sf[CH * 256];                // 64 KB
    const int b   = blockIdx.x >> 3;
    const int n0  = (blockIdx.x & 7) * 256;
    const int tid = threadIdx.x;

    const float* fb = fea2 + (size_t)b * CH * NPT + n0;
    for (int c = 0; c < CH; ++c) sf[c * 256 + tid] = fb[c * NPT + tid];
    __syncthreads();

    const int n = n0 + tid;
    _Float16* gr = gT + ((size_t)b * NPT + n) * 32;
    _Float16* fr = fT + ((size_t)b * NPT + n) * 32;

#pragma unroll 1
    for (int o = 0; o < 16; ++o) {
        float af = bf[o], ag = bg[o];
#pragma unroll
        for (int c = 0; c < CH; ++c) {
            float x = sf[c * 256 + tid];
            af += Wf[o * CH + c] * x;
            ag += Wg[o * CH + c] * x;
        }
        fr[o]      = (_Float16)fmaxf(af, 0.f);
        gr[o]      = (_Float16)fmaxf(ag, 0.f);
        fr[16 + o] = (_Float16)0.f;               // K-pad
        gr[16 + o] = (_Float16)0.f;
    }

    // interleaved key position inside the 32-block
    const int blk = n & ~31;
    const int j   = n & 31;
    const int p   = (j < 16) ? (2 * j) : (2 * (j - 16) + 1);

    _Float16* hb = h16 + (size_t)b * CH * NPT;
#pragma unroll 1
    for (int o = 0; o < CH; ++o) {
        float ah = bh[o];
#pragma unroll
        for (int c = 0; c < CH; ++c) ah += Wh[o * CH + c] * sf[c * 256 + tid];
        hb[(size_t)o * NPT + blk + p] = (_Float16)fmaxf(ah, 0.f);
    }
}

// ---------------------------------------------------------------------------
// Kernel 3: fused attention + residual + MLP epilogue. One wave32 per
// (batch, 16-query tile).
//
// Max-free softmax (logits are O(1): relu'd 0.1-scale projections, K=16 ->
// exp cannot overflow; softmax is shift-invariant -> exact). Row sums ride
// the matrix pipe: accL = P x ones. P pairs packed -> 8 ds_store_b32/block.
// 7 v_wmma_f32_16x16x32_f16 per 32-key block; MLP epilogue on
// v_wmma_f32_16x16x4_f32 (64 WMMAs replace ~2k per-lane VMEM+FMA).
// ---------------------------------------------------------------------------
__global__ __launch_bounds__(32) void k3_attn(const float* __restrict__ fea2,
                                              const _Float16* __restrict__ gT,
                                              const _Float16* __restrict__ fT,
                                              const _Float16* __restrict__ h16,
                                              const float* __restrict__ gamma,
                                              const float* __restrict__ W1, const float* __restrict__ b1,
                                              const float* __restrict__ W2, const float* __restrict__ b2,
                                              float* __restrict__ out) {
    __shared__ __align__(64) _Float16 pt[2][16 * 32]; // double-buffered P tile
    __shared__ float xb[CH * 16];                     // x = gamma*o + fea2

    const int b    = blockIdx.x >> 7;
    const int n0   = (blockIdx.x & 127) * 16;
    const int lane = threadIdx.x;
    const int lo   = lane & 15;
    const int hi   = lane >> 4;

    // ---- A operand: Q = g tile (16 x 32), contiguous k = 16*hi + s (matches B)
    const _Float16* gq = gT + ((size_t)b * NPT + n0) * 32;
    V16 Ag;
    Ag.h[0] = *(const v8h*)(gq + lo * 32 + hi * 16);
    Ag.h[1] = *(const v8h*)(gq + lo * 32 + hi * 16 + 8);

    // all-ones B operand for row-sum WMMA
    v16h ones;
#pragma unroll
    for (int i = 0; i < 16; ++i) ones[i] = (_Float16)1.0f;

    v8f acc0 = {}, acc1 = {}, acc2 = {}, acc3 = {}, accL = {};

    const _Float16* fk = fT  + (size_t)b * NPT * 32;
    const _Float16* hv = h16 + (size_t)b * CH * NPT;

#pragma unroll 1
    for (int m0 = 0; m0 < NPT; m0 += 32) {
        // ---- S = Q * K^T for 32 keys (two 16-key WMMA tiles)
        v16h Bf0 = *(const v16h*)(fk + (size_t)(m0 + lo) * 32 + hi * 16);
        v16h Bf1 = *(const v16h*)(fk + (size_t)(m0 + 16 + lo) * 32 + hi * 16);
        v8f s0 = {}, s1 = {};
        s0 = __builtin_amdgcn_wmma_f32_16x16x32_f16(false, Ag.v, false, Bf0, (short)0, s0, false, false);
        s1 = __builtin_amdgcn_wmma_f32_16x16x32_f16(false, Ag.v, false, Bf1, (short)0, s1, false, false);

        // ---- P = exp(S), packed pair store (key j -> pos 2j, key 16+j -> 2j+1)
        _Float16* pb = pt[(m0 >> 5) & 1];
#pragma unroll
        for (int r = 0; r < 8; ++r) {
            v2h pr;
            pr[0] = (_Float16)__expf(s0[r]);
            pr[1] = (_Float16)__expf(s1[r]);
            *(v2h*)(pb + (r + 8 * hi) * 32 + 2 * lo) = pr;
        }
        __syncthreads();   // s_wait_dscnt + (single-wave) NOP barrier

        // ---- C-layout -> A-layout reload of P (position order)
        V16 Ap;
        Ap.h[0] = *(const v8h*)(pb + lo * 32 + hi * 16);
        Ap.h[1] = *(const v8h*)(pb + lo * 32 + hi * 16 + 8);

        // ---- acc += P * V ; accL += P * 1   (V rows pre-interleaved in k2)
        v16h Bv0 = *(const v16h*)(hv + (size_t)(0  + lo) * NPT + m0 + hi * 16);
        v16h Bv1 = *(const v16h*)(hv + (size_t)(16 + lo) * NPT + m0 + hi * 16);
        v16h Bv2 = *(const v16h*)(hv + (size_t)(32 + lo) * NPT + m0 + hi * 16);
        v16h Bv3 = *(const v16h*)(hv + (size_t)(48 + lo) * NPT + m0 + hi * 16);
        acc0 = __builtin_amdgcn_wmma_f32_16x16x32_f16(false, Ap.v, false, Bv0, (short)0, acc0, false, false);
        acc1 = __builtin_amdgcn_wmma_f32_16x16x32_f16(false, Ap.v, false, Bv1, (short)0, acc1, false, false);
        acc2 = __builtin_amdgcn_wmma_f32_16x16x32_f16(false, Ap.v, false, Bv2, (short)0, acc2, false, false);
        acc3 = __builtin_amdgcn_wmma_f32_16x16x32_f16(false, Ap.v, false, Bv3, (short)0, acc3, false, false);
        accL = __builtin_amdgcn_wmma_f32_16x16x32_f16(false, Ap.v, false, ones, (short)0, accL, false, false);
    }

    // ---- x = gamma * (acc / rowsum) + fea2  -> LDS
    const float gm = gamma[0];
    const float* f2 = fea2 + (size_t)b * CH * NPT;
#pragma unroll
    for (int r = 0; r < 8; ++r) {
        const int nl = r + 8 * hi;
        const float il = 1.0f / accL[r];          // every column of accL = rowsum
        xb[(0  + lo) * 16 + nl] = gm * acc0[r] * il + f2[(size_t)(0  + lo) * NPT + n0 + nl];
        xb[(16 + lo) * 16 + nl] = gm * acc1[r] * il + f2[(size_t)(16 + lo) * NPT + n0 + nl];
        xb[(32 + lo) * 16 + nl] = gm * acc2[r] * il + f2[(size_t)(32 + lo) * NPT + n0 + nl];
        xb[(48 + lo) * 16 + nl] = gm * acc3[r] * il + f2[(size_t)(48 + lo) * NPT + n0 + nl];
    }
    __syncthreads();

    float po0 = 0.f, po1 = 0.f, po2 = 0.f;

#if __has_builtin(__builtin_amdgcn_wmma_f32_16x16x4_f32)
    // ---- y = relu(W1 x + b1) on the matrix pipe: D[o,n] = sum_c W1[o,c] x[c,n]
    // A (16x4 f32): lane -> M=lo, K=2*hi+{0,1}  => contiguous v2f from W1 rows
    // B (4x16 f32): lane -> N=lo, K=2*hi+{0,1}  => two b32 LDS reads of x
    v8f y0 = {}, y1 = {}, y2 = {}, y3 = {};
#pragma unroll
    for (int kk = 0; kk < 16; ++kk) {
        const int k0 = kk * 4;
        v2f bx;
        bx[0] = xb[(k0 + 2 * hi) * 16 + lo];
        bx[1] = xb[(k0 + 2 * hi + 1) * 16 + lo];
        v2f a0 = *(const v2f*)(W1 + (size_t)(0  + lo) * CH + k0 + 2 * hi);
        v2f a1 = *(const v2f*)(W1 + (size_t)(16 + lo) * CH + k0 + 2 * hi);
        v2f a2 = *(const v2f*)(W1 + (size_t)(32 + lo) * CH + k0 + 2 * hi);
        v2f a3 = *(const v2f*)(W1 + (size_t)(48 + lo) * CH + k0 + 2 * hi);
        y0 = __builtin_amdgcn_wmma_f32_16x16x4_f32(false, a0, false, bx, (short)0, y0, false, false);
        y1 = __builtin_amdgcn_wmma_f32_16x16x4_f32(false, a1, false, bx, (short)0, y1, false, false);
        y2 = __builtin_amdgcn_wmma_f32_16x16x4_f32(false, a2, false, bx, (short)0, y2, false, false);
        y3 = __builtin_amdgcn_wmma_f32_16x16x4_f32(false, a3, false, bx, (short)0, y3, false, false);
    }
    // bias + relu + W2 reduction; lane owns y[o = 16t + r + 8*hi][n = lo]
#pragma unroll
    for (int r = 0; r < 8; ++r) {
        const int o0 = r + 8 * hi;
        float yv;
        yv = fmaxf(y0[r] + b1[o0],      0.f); po0 += W2[o0]       * yv; po1 += W2[64 + o0]      * yv; po2 += W2[128 + o0]      * yv;
        yv = fmaxf(y1[r] + b1[o0 + 16], 0.f); po0 += W2[o0 + 16]  * yv; po1 += W2[64 + o0 + 16] * yv; po2 += W2[128 + o0 + 16] * yv;
        yv = fmaxf(y2[r] + b1[o0 + 32], 0.f); po0 += W2[o0 + 32]  * yv; po1 += W2[64 + o0 + 32] * yv; po2 += W2[128 + o0 + 32] * yv;
        yv = fmaxf(y3[r] + b1[o0 + 48], 0.f); po0 += W2[o0 + 48]  * yv; po1 += W2[64 + o0 + 48] * yv; po2 += W2[128 + o0 + 48] * yv;
    }
#else
    // fallback: VALU MLP
    float xcol[CH];
#pragma unroll
    for (int c = 0; c < CH; ++c) xcol[c] = xb[c * 16 + lo];
    const int obase = hi * 32;
#pragma unroll 1
    for (int oo = 0; oo < 32; ++oo) {
        const int o1 = obase + oo;
        float a = b1[o1];
#pragma unroll
        for (int c = 0; c < CH; ++c) a += W1[o1 * CH + c] * xcol[c];
        const float yv = fmaxf(a, 0.f);
        po0 += W2[o1] * yv; po1 += W2[64 + o1] * yv; po2 += W2[128 + o1] * yv;
    }
#endif

    po0 += __shfl_xor(po0, 16, 32);
    po1 += __shfl_xor(po1, 16, 32);
    po2 += __shfl_xor(po2, 16, 32);

    if (hi == 0) {
        float* ob = out + (size_t)b * 3 * NPT + n0 + lo;
        ob[0]        = po0 + b2[0];
        ob[NPT]      = po1 + b2[1];
        ob[2 * NPT]  = po2 + b2[2];
    }
}

// ---------------------------------------------------------------------------
extern "C" void kernel_launch(void* const* d_in, const int* in_sizes, int n_in,
                              void* d_out, int out_size, void* d_ws, size_t ws_size,
                              hipStream_t stream) {
    const float* feature = (const float*)d_in[0];
    const float* xyz     = (const float*)d_in[1];
    const float* Wf      = (const float*)d_in[2];
    const float* bf      = (const float*)d_in[3];
    const float* Wg      = (const float*)d_in[4];
    const float* bg      = (const float*)d_in[5];
    const float* Wh      = (const float*)d_in[6];
    const float* bh      = (const float*)d_in[7];
    const float* gamma   = (const float*)d_in[8];
    const float* W1      = (const float*)d_in[9];
    const float* b1      = (const float*)d_in[10];
    const float* W2      = (const float*)d_in[11];
    const float* b2      = (const float*)d_in[12];

    char* ws = (char*)d_ws;
    float*    fea2  = (float*)ws;                         // [0,4MB)
    float*    featT = (float*)(ws + 4194304);             // [4MB,8MB) - dead after k1
    _Float16* gT    = (_Float16*)(ws + 4194304);          // aliases featT region
    _Float16* fT    = (_Float16*)(ws + 5242880);
    _Float16* h16   = (_Float16*)(ws + 6291456);

    float* out = (float*)d_out;

    k0_tr  <<<BATCH * (NPT / 64),  256, 0, stream>>>(feature, featT);
    k1_knn <<<BATCH * (NPT / 256), 256, 0, stream>>>(xyz, featT, fea2);
    k2_fgh <<<BATCH * (NPT / 256), 256, 0, stream>>>(fea2, Wf, bf, Wg, bg, Wh, bh, gT, fT, h16);
    k3_attn<<<BATCH * (NPT / 16),  32,  0, stream>>>(fea2, gT, fT, h16, gamma, W1, b1, W2, b2, out);
}
